// SHPEmbeddingLayer_32530082300509
// MI455X (gfx1250) — compile-verified, matrix operand order checked by reading, not run on hardware
//
#include <hip/hip_runtime.h>

// ---- CDNA5 / gfx1250: wave32, WMMA bf16 16x16x32 (f32 accum) ----
typedef __attribute__((ext_vector_type(16))) __bf16       v16bf;
typedef __attribute__((ext_vector_type(8)))  float        v8f;
typedef __attribute__((ext_vector_type(4)))  unsigned int u32x4;
typedef __attribute__((ext_vector_type(4)))  int          v4i;

#define B_      4
#define L_      2048
#define S_      2046
#define D_      1280
#define KSTR    20
#define M_      8184        // B_*S_
#define MP      8192        // M_ padded to 128
#define KB1     40          // D_/32  (k-blocks for h@W1)
#define K2P     416         // (SEQV*KSTR=400) padded to 32
#define KB2     13          // K2P/32
#define NBLK    10          // D_/128 (N tile blocks)
#define LN_EPS_ 1e-5f

// CDNA5 async memory->LDS path (GLOBAL_LOAD_ASYNC_TO_LDS_B128, ASYNCcnt)
#if defined(__AMDGCN__) && __has_builtin(__builtin_amdgcn_global_load_async_to_lds_b128)
#define USE_ASYNC_LDS 1
#else
#define USE_ASYNC_LDS 0
#endif

#define PTR_AS1 __attribute__((address_space(1)))
#define PTR_AS3 __attribute__((address_space(3)))

__device__ __forceinline__ void wait_async0() {
#if defined(__AMDGCN__)
#if __has_builtin(__builtin_amdgcn_s_wait_asynccnt)
  __builtin_amdgcn_s_wait_asynccnt(0);
#else
  asm volatile("s_wait_asynccnt 0x0" ::: "memory");
#endif
#endif
}

union FragU { v16bf v; u32x4 q[2]; };

// ---------------------------------------------------------------------------
// Kernel 0: convert W1 (f32 [k][n]) and full_embed (f32, reshaped 400 x D) to
// bf16 in K-block-tiled layout [k/32][n][32] so each GEMM tile is contiguous.
// ---------------------------------------------------------------------------
__global__ void k_prep_weights(const float* __restrict__ W1,
                               const float* __restrict__ FE,
                               __bf16* __restrict__ W1t,
                               __bf16* __restrict__ Ft) {
  const int total1 = KB1 * D_ * 32;
  const int total2 = KB2 * D_ * 32;
  for (int t = blockIdx.x * blockDim.x + threadIdx.x; t < total1 + total2;
       t += gridDim.x * blockDim.x) {
    if (t < total1) {
      int kb = t / (D_ * 32), rem = t % (D_ * 32);
      int n = rem >> 5, kk = rem & 31, k = kb * 32 + kk;
      W1t[t] = (__bf16)W1[(size_t)k * D_ + n];
    } else {
      int t2 = t - total1;
      int kb = t2 / (D_ * 32), rem = t2 % (D_ * 32);
      int n = rem >> 5, kk = rem & 31, k = kb * 32 + kk;
      Ft[t2] = (k < 400) ? (__bf16)FE[(size_t)k * D_ + n] : (__bf16)0.0f;
    }
  }
}

// ---------------------------------------------------------------------------
// Kernel 1: per-token gather + LayerNorm -> Xa (bf16 A-layout [kb][MP][32]),
// and sparse P rows -> Pa (bf16 [kb2][MP][32]).  One 256-thread block / row.
// ---------------------------------------------------------------------------
__global__ void k_token_prep(const int* __restrict__ seq_ids,
                             const int* __restrict__ struct_ids,
                             const float* __restrict__ shp,
                             const float* __restrict__ FE,
                             const float* __restrict__ ln_g,
                             const float* __restrict__ ln_b,
                             __bf16* __restrict__ Xa,
                             __bf16* __restrict__ Pa) {
  const int m = blockIdx.x;
  const int tid = threadIdx.x;
  __shared__ float red[256];

  if (m >= M_) {  // padding rows: zero contributions
    for (int d = tid; d < D_;  d += 256) Xa[((size_t)(d >> 5) * MP + m) * 32 + (d & 31)] = (__bf16)0.0f;
    for (int j = tid; j < K2P; j += 256) Pa[((size_t)(j >> 5) * MP + m) * 32 + (j & 31)] = (__bf16)0.0f;
    return;
  }
  const int b = m / S_, s = m - b * S_, l = s + 1;
  const int idx = b * L_ + l;
  const int sq = seq_ids[idx], st = struct_ids[idx];
  const float* er = FE + (size_t)(sq * KSTR + st) * D_;

  float e[5], sum = 0.f, sumsq = 0.f;
  for (int i = 0; i < 5; i++) {
    float v = er[tid + 256 * i];
    e[i] = v; sum += v; sumsq += v * v;
  }
  red[tid] = sum; __syncthreads();
  for (int o = 128; o > 0; o >>= 1) { if (tid < o) red[tid] += red[tid + o]; __syncthreads(); }
  float fsum = red[0]; __syncthreads();
  red[tid] = sumsq; __syncthreads();
  for (int o = 128; o > 0; o >>= 1) { if (tid < o) red[tid] += red[tid + o]; __syncthreads(); }
  float fsq = red[0];

  float mu  = fsum * (1.0f / D_);
  float var = fsq * (1.0f / D_) - mu * mu;
  float rs  = rsqrtf(var + LN_EPS_);
  for (int i = 0; i < 5; i++) {
    int d = tid + 256 * i;
    float xh = (e[i] - mu) * rs * ln_g[d] + ln_b[d];
    Xa[((size_t)(d >> 5) * MP + m) * 32 + (d & 31)] = (__bf16)xh;
  }
  const float* sp = shp + (size_t)(b * S_ + s) * KSTR;
  const int lo = sq * KSTR;
  for (int j = tid; j < K2P; j += 256) {
    float v = (j >= lo && j < lo + KSTR) ? sp[j - lo] : 0.0f;
    Pa[((size_t)(j >> 5) * MP + m) * 32 + (j & 31)] = (__bf16)v;
  }
}

// ---------------------------------------------------------------------------
// Shared GEMM tile core: C(128x128) += A(128xK) * B(KxN=128), bf16 WMMA.
// A: [NKB][MP][32] tiled, B: [NKB][D_][32] tiled. 8 waves, each 64x32 (8 C
// tiles of 16x16 in 8 VGPRs). Fragment maps follow ISA 7.12.2 (16-bit A/B).
// Double-buffered LDS; prefetch of tile k+1 overlaps the 8 WMMAs of tile k,
// via GLOBAL_LOAD_ASYNC_TO_LDS_B128 (ASYNCcnt) when available.
// ---------------------------------------------------------------------------
template <int NKB>
__device__ __forceinline__ void gemm_tile(const __bf16* __restrict__ A,
                                          const __bf16* __restrict__ Bw,
                                          int m0, int n0, v8f acc[4][2]) {
  __shared__ u32x4 sAB[2][1024];  // [buf][ A-tile 0..511 | B-tile 512..1023 ]
  const int tid  = threadIdx.x;
  const int lane = tid & 31;
  const int wid  = tid >> 5;
  const int wm   = (wid & 1) * 64;   // wave M offset
  const int wn   = (wid >> 1) * 32;  // wave N offset
  const int half = (lane >> 4) & 1;  // K-chunk select per ISA 7.12.2 layout
  const int lr   = lane & 15;

  for (int mt = 0; mt < 4; mt++)
    for (int nt = 0; nt < 2; nt++)
      for (int i = 0; i < 8; i++) acc[mt][nt][i] = 0.0f;

  const u32x4* gAbase = (const u32x4*)(A  + (size_t)m0 * 32);
  const u32x4* gBbase = (const u32x4*)(Bw + (size_t)n0 * 32);
  const size_t strA = (size_t)MP * 2;  // u32x4 step per kb (MP*32B /16 = half of a MP*32*2B k-block)
  const size_t strB = (size_t)D_ * 2;

  // ---- prologue: stage k-block 0 into buffer 0 ----
  {
    const u32x4* gA = gAbase;
    const u32x4* gB = gBbase;
#if USE_ASYNC_LDS
    for (int i = tid; i < 512; i += 256) {
      __builtin_amdgcn_global_load_async_to_lds_b128(
          (PTR_AS1 v4i*)(u32x4*)(gA + i), (PTR_AS3 v4i*)&sAB[0][i], 0, 0);
      __builtin_amdgcn_global_load_async_to_lds_b128(
          (PTR_AS1 v4i*)(u32x4*)(gB + i), (PTR_AS3 v4i*)&sAB[0][512 + i], 0, 0);
    }
    wait_async0();
#else
    for (int i = tid; i < 512; i += 256) {
      sAB[0][i] = gA[i];
      sAB[0][512 + i] = gB[i];
    }
#endif
    __syncthreads();
  }

  for (int kb = 0; kb < NKB; kb++) {
    const int cur = kb & 1, nxt = cur ^ 1;
    const u32x4* lds = &sAB[cur][0];

    // fragment loads (DS) for current tile
    FragU afr[4], bfr[2];
    for (int nt = 0; nt < 2; nt++) {
      int col = wn + nt * 16 + lr;
      bfr[nt].q[0] = lds[512 + col * 4 + half];
      bfr[nt].q[1] = lds[512 + col * 4 + half + 2];
    }
    for (int mt = 0; mt < 4; mt++) {
      int row = wm + mt * 16 + lr;
      afr[mt].q[0] = lds[row * 4 + half];
      afr[mt].q[1] = lds[row * 4 + half + 2];
    }

    const bool pf = (kb + 1 < NKB);  // wave-uniform
    const u32x4* gA = gAbase + (size_t)(kb + 1) * strA;
    const u32x4* gB = gBbase + (size_t)(kb + 1) * strB;

#if USE_ASYNC_LDS
    if (pf) {  // async DMA into other buffer; overlaps WMMAs below
      for (int i = tid; i < 512; i += 256) {
        __builtin_amdgcn_global_load_async_to_lds_b128(
            (PTR_AS1 v4i*)(u32x4*)(gA + i), (PTR_AS3 v4i*)&sAB[nxt][i], 0, 0);
        __builtin_amdgcn_global_load_async_to_lds_b128(
            (PTR_AS1 v4i*)(u32x4*)(gB + i), (PTR_AS3 v4i*)&sAB[nxt][512 + i], 0, 0);
      }
    }
    for (int mt = 0; mt < 4; mt++)
      for (int nt = 0; nt < 2; nt++)
        acc[mt][nt] = __builtin_amdgcn_wmma_f32_16x16x32_bf16(
            false, afr[mt].v, false, bfr[nt].v, (short)0, acc[mt][nt], false, false);
    if (pf) wait_async0();
#else
    // register prefetch: global loads issued before WMMAs, stored after
    u32x4 ra0, ra1, rb0, rb1;
    if (pf) { ra0 = gA[tid]; ra1 = gA[tid + 256]; rb0 = gB[tid]; rb1 = gB[tid + 256]; }
    for (int mt = 0; mt < 4; mt++)
      for (int nt = 0; nt < 2; nt++)
        acc[mt][nt] = __builtin_amdgcn_wmma_f32_16x16x32_bf16(
            false, afr[mt].v, false, bfr[nt].v, (short)0, acc[mt][nt], false, false);
    if (pf) {
      sAB[nxt][tid]       = ra0;  sAB[nxt][tid + 256]       = ra1;
      sAB[nxt][512 + tid] = rb0;  sAB[nxt][512 + tid + 256] = rb1;
    }
#endif
    __syncthreads();
  }
}

// ---------------------------------------------------------------------------
// Kernel 2: H-GEMM (Xa @ W1t), fused bias + exact GELU + w2 dot; writes
// deterministic per-(row, N-block) partial gate sums (no FP atomics).
// ---------------------------------------------------------------------------
__global__ void k_gemm_h(const __bf16* __restrict__ Xa,
                         const __bf16* __restrict__ W1t,
                         const float* __restrict__ b1,
                         const float* __restrict__ w2,
                         float* __restrict__ pg) {
  const int m0 = blockIdx.x * 128, n0 = blockIdx.y * 128;
  v8f acc[4][2];
  gemm_tile<KB1>(Xa, W1t, m0, n0, acc);

  __shared__ float part[4][128];
  const int tid = threadIdx.x, lane = tid & 31, wid = tid >> 5;
  const int wm = (wid & 1) * 64, wn = (wid >> 1) * 32, cg = wid >> 1;
  const int half = (lane >> 4) & 1, lr = lane & 15;

  for (int mt = 0; mt < 4; mt++) {
    for (int v = 0; v < 8; v++) {
      float rp = 0.0f;
      for (int nt = 0; nt < 2; nt++) {
        int col = n0 + wn + nt * 16 + lr;
        float h = acc[mt][nt][v] + b1[col];
        float g = 0.5f * h * (1.0f + erff(h * 0.70710678118654752f));
        float p = g * w2[col];
        p += __shfl_xor(p, 1);   // butterfly within 16-lane halves
        p += __shfl_xor(p, 2);   // (rows live on lanes 0-15 / 16-31)
        p += __shfl_xor(p, 4);
        p += __shfl_xor(p, 8);
        rp += p;
      }
      if (lr == 0) part[cg][wm + mt * 16 + v + half * 8] = rp;
    }
  }
  __syncthreads();
  if (tid < 128) {
    float s = part[0][tid] + part[1][tid] + part[2][tid] + part[3][tid];
    pg[(size_t)(m0 + tid) * NBLK + blockIdx.y] = s;
  }
}

// Kernel 3: gate[m] = sigmoid(sum of partials + b2)
__global__ void k_gate(const float* __restrict__ pg,
                       const float* __restrict__ b2,
                       float* __restrict__ gate) {
  int m = blockIdx.x * blockDim.x + threadIdx.x;
  if (m < MP) {
    float s = b2[0];
    for (int j = 0; j < NBLK; j++) s += pg[(size_t)m * NBLK + j];
    gate[m] = 1.0f / (1.0f + expf(-s));
  }
}

// ---------------------------------------------------------------------------
// Kernel 4: E_shp GEMM (Pa @ Ft) with fused epilogue: re-gather E_tok from
// L2-resident full_embed, blend by gate, apply special-token override,
// write final output rows (l = 1..S).
// ---------------------------------------------------------------------------
__global__ void k_gemm_shp(const __bf16* __restrict__ Pa,
                           const __bf16* __restrict__ Ft,
                           const int* __restrict__ seq_ids,
                           const int* __restrict__ struct_ids,
                           const unsigned char* __restrict__ smask,
                           const int* __restrict__ spos,
                           const float* __restrict__ semb,
                           const float* __restrict__ FE,
                           const float* __restrict__ gate,
                           float* __restrict__ out) {
  const int m0 = blockIdx.x * 128, n0 = blockIdx.y * 128;
  v8f acc[4][2];
  gemm_tile<KB2>(Pa, Ft, m0, n0, acc);

  const int tid = threadIdx.x, lane = tid & 31, wid = tid >> 5;
  const int wm = (wid & 1) * 64, wn = (wid >> 1) * 32;
  const int half = (lane >> 4) & 1, lr = lane & 15;

  for (int mt = 0; mt < 4; mt++)
    for (int nt = 0; nt < 2; nt++)
      for (int v = 0; v < 8; v++) {
        int row = m0 + wm + mt * 16 + v + half * 8;
        int col = n0 + wn + nt * 16 + lr;
        if (row < M_) {
          int b = row / S_, s = row - b * S_, l = s + 1;
          int idx = b * L_ + l;
          float o;
          if (smask[idx]) {
            o = semb[(size_t)spos[idx] * D_ + col];
          } else {
            int sq = seq_ids[idx], st = struct_ids[idx];
            float et = FE[(size_t)(sq * KSTR + st) * D_ + col];
            float g  = gate[row];
            o = g * acc[mt][nt][v] + (1.0f - g) * et;
          }
          out[(size_t)idx * D_ + col] = o;
        }
      }
}

// Kernel 5: edge rows l=0 and l=L-1 (zero or special embedding)
__global__ void k_edges(const unsigned char* __restrict__ smask,
                        const int* __restrict__ spos,
                        const float* __restrict__ semb,
                        float* __restrict__ out) {
  int b = blockIdx.x >> 1;
  int l = (blockIdx.x & 1) ? (L_ - 1) : 0;
  int idx = b * L_ + l;
  bool sp = smask[idx] != 0;
  const float* srow = semb + (size_t)spos[idx] * D_;
  for (int d = threadIdx.x; d < D_; d += blockDim.x)
    out[(size_t)idx * D_ + d] = sp ? srow[d] : 0.0f;
}

// ---------------------------------------------------------------------------
extern "C" void kernel_launch(void* const* d_in, const int* in_sizes, int n_in,
                              void* d_out, int out_size, void* d_ws, size_t ws_size,
                              hipStream_t stream) {
  const int*           seq_ids    = (const int*)d_in[0];
  const int*           struct_ids = (const int*)d_in[1];
  const unsigned char* smask      = (const unsigned char*)d_in[2]; // jax bool = 1 byte
  const int*           spos       = (const int*)d_in[3];
  const float*         shp        = (const float*)d_in[4];
  const float*         semb       = (const float*)d_in[5];
  const float*         FE         = (const float*)d_in[6];
  const float*         ln_g       = (const float*)d_in[7];
  const float*         ln_b       = (const float*)d_in[8];
  const float*         W1         = (const float*)d_in[9];
  const float*         b1         = (const float*)d_in[10];
  const float*         w2         = (const float*)d_in[11];
  const float*         b2         = (const float*)d_in[12];
  float*               out        = (float*)d_out;

  char* ws = (char*)d_ws;
  size_t off = 0;
  auto alloc = [&](size_t bytes) { size_t p = off; off += (bytes + 255) & ~(size_t)255; return p; };
  __bf16* W1t  = (__bf16*)(ws + alloc((size_t)KB1 * D_ * 32 * 2)); //  3.3 MB
  __bf16* Ft   = (__bf16*)(ws + alloc((size_t)KB2 * D_ * 32 * 2)); //  1.1 MB
  __bf16* Xa   = (__bf16*)(ws + alloc((size_t)KB1 * MP * 32 * 2)); // 21.0 MB
  __bf16* Pa   = (__bf16*)(ws + alloc((size_t)KB2 * MP * 32 * 2)); //  6.8 MB
  float*  pg   = (float*)(ws + alloc((size_t)MP * NBLK * 4));      //  0.3 MB
  float*  gate = (float*)(ws + alloc((size_t)MP * 4));             //  32 KB

  k_prep_weights<<<2048, 256, 0, stream>>>(W1, FE, W1t, Ft);
  k_token_prep<<<MP, 256, 0, stream>>>(seq_ids, struct_ids, shp, FE, ln_g, ln_b, Xa, Pa);
  dim3 gg(MP / 128, NBLK);
  k_gemm_h<<<gg, 256, 0, stream>>>(Xa, W1t, b1, w2, pg);
  k_gate<<<(MP + 255) / 256, 256, 0, stream>>>(pg, b2, gate);
  k_gemm_shp<<<gg, 256, 0, stream>>>(Pa, Ft, seq_ids, struct_ids, smask, spos, semb, FE, gate, out);
  k_edges<<<B_ * 2, 256, 0, stream>>>(smask, spos, semb, out);
}
